// FASTMultiHeadAttention_71940702208678
// MI455X (gfx1250) — matrix operand order repaired and checked by reading, not compiled, not run on hardware
//
#include <hip/hip_runtime.h>

// ---------------------------------------------------------------------------
// fastmax (Taylor-linearized causal attention with RPE bias) for gfx1250.
// b=4, h=16, n=1024, d=64, f32 in/out. bf16 WMMA (v_wmma_f32_16x16x32_bf16)
// for QK^T, the RPE band-bias GEMM, and P@V; f32 accumulation throughout.
// ---------------------------------------------------------------------------

typedef __attribute__((ext_vector_type(16))) __bf16 v16bf;
typedef __attribute__((ext_vector_type(8)))  __bf16 v8bf;
typedef __attribute__((ext_vector_type(8)))  float  v8f;

#define WMMA_BF16(a, b, c) \
  __builtin_amdgcn_wmma_f32_16x16x32_bf16(false, (a), false, (b), (short)0, (c), false, false)

__device__ __forceinline__ v16bf cat16(v8bf lo, v8bf hi) {
  return __builtin_shufflevector(lo, hi, 0,1,2,3,4,5,6,7,8,9,10,11,12,13,14,15);
}

__device__ __forceinline__ v8bf pack8(const float* f) {
  v8bf p;
#pragma unroll
  for (int e = 0; e < 8; ++e) p[e] = (__bf16)f[e];
  return p;
}

namespace {
constexpr int   NSEQ = 1024;
constexpr int   DIM  = 64;
constexpr float DROP = 0.1f;

// LDS layout (bytes). Q tile (8KB) overlaps the E scratch region (20.25KB):
// Q is consumed into registers before the first E write (extra barrier).
constexpr int OFF_E = 0;       // 4 waves * 16 * 81 f32 = 20736  (Q bf16 64x64 overlaps)
constexpr int OFF_K = 20736;   // 64 x 64 bf16 = 8192
constexpr int OFF_V = 28928;   // Vt: 64(d) x 64(key) bf16 = 8192
constexpr int OFF_R = 37120;   // 128 x 64 bf16 = 16384
constexpr int OFF_P = 53504;   // 4 waves * 16 x 64 bf16 = 8192
constexpr int SMEM_BYTES = 61696;
}

__launch_bounds__(128, 1)
__global__ void fastmax_fwd_kernel(const float* __restrict__ gq,
                                   const float* __restrict__ gk,
                                   const float* __restrict__ gv,
                                   const float* __restrict__ gnz,
                                   const float* __restrict__ grpe,
                                   float* __restrict__ gout)
{
  __shared__ __align__(16) char smem[SMEM_BYTES];
  float*  Elds = (float *)(smem + OFF_E);   // per-wave 16x81 f32 bias scratch
  __bf16* Qlds = (__bf16*)(smem + OFF_E);   // 64x64 bf16 (prologue only)
  __bf16* Klds = (__bf16*)(smem + OFF_K);   // key tile, row-major [key][d]
  __bf16* Vt   = (__bf16*)(smem + OFF_V);   // value tile transposed [d][key]
  __bf16* Rlds = (__bf16*)(smem + OFF_R);   // rpe window, row-major [rel][d]
  __bf16* Plds = (__bf16*)(smem + OFF_P);   // per-wave 16x64 bf16 scores

  const int tile = blockIdx.x;          // row tile, 0..15
  const int bh   = blockIdx.y;          // 0..63
  const int i0   = tile * 64;
  const int tid  = threadIdx.x;         // 0..127
  const int wv   = tid >> 5;            // wave 0..3
  const int lane = tid & 31;
  const int hf   = lane >> 4;           // lane half (0/1)
  const int lc   = lane & 15;           // lane within half

  const size_t hoff = (size_t)bh * NSEQ * DIM;
  const float* Q  = gq  + hoff;
  const float* K  = gk  + hoff;
  const float* V  = gv  + hoff;
  const float* NZ = gnz + hoff;
  float*       O  = gout + hoff;

  // ---- prologue: stage l2-normalized Q tile as bf16 ----
  {
    const int row = tid >> 1;
    const int hc  = (tid & 1) * 32;
    const float* src = Q + (size_t)(i0 + row) * DIM + hc;
    float vals[32];
    float ss = 0.f;
#pragma unroll
    for (int c = 0; c < 32; c += 4) {
      float4 f = *(const float4*)(src + c);
      vals[c] = f.x; vals[c+1] = f.y; vals[c+2] = f.z; vals[c+3] = f.w;
      ss += f.x*f.x + f.y*f.y + f.z*f.z + f.w*f.w;
    }
    ss += __shfl_xor(ss, 1);            // pair (row-half) reduce, wave32
    const float rs = rsqrtf(ss);
#pragma unroll
    for (int c = 0; c < 32; ++c) vals[c] *= rs;
#pragma unroll
    for (int c = 0; c < 32; c += 8)
      *(v8bf*)(Qlds + row * DIM + hc + c) = pack8(vals + c);
  }
  __syncthreads();

  // capture per-wave Q A-fragments (16 rows x K=64 -> two K=32 frags)
  v16bf qA0, qA1;
  {
    const __bf16* base = Qlds + (wv * 16 + lc) * DIM + 16 * hf;
    qA0 = cat16(*(const v8bf*)(base),      *(const v8bf*)(base + 8));
    qA1 = cat16(*(const v8bf*)(base + 32), *(const v8bf*)(base + 40));
  }
  __syncthreads();                      // Qlds region becomes Elds scratch

  v8f accO[4];
#pragma unroll
  for (int ct = 0; ct < 4; ++ct) accO[ct] = (v8f)0.0f;
  float dsum[8] = {0.f,0.f,0.f,0.f,0.f,0.f,0.f,0.f};

  const int ntile = tile + 1;           // causal: only j-tiles <= i-tile
  for (int jt = 0; jt < ntile; ++jt) {
    const int j0 = jt * 64;

    // ---- stage K tile: l2-normalize rows, bf16 row-major ----
    {
      const int row = tid >> 1;
      const int hc  = (tid & 1) * 32;
      const float* src = K + (size_t)(j0 + row) * DIM + hc;
      float vals[32];
      float ss = 0.f;
#pragma unroll
      for (int c = 0; c < 32; c += 4) {
        float4 f = *(const float4*)(src + c);
        vals[c] = f.x; vals[c+1] = f.y; vals[c+2] = f.z; vals[c+3] = f.w;
        ss += f.x*f.x + f.y*f.y + f.z*f.z + f.w*f.w;
      }
      ss += __shfl_xor(ss, 1);
      const float rs = rsqrtf(ss);
#pragma unroll
      for (int c = 0; c < 32; ++c) vals[c] *= rs;
#pragma unroll
      for (int c = 0; c < 32; c += 8)
        *(v8bf*)(Klds + row * DIM + hc + c) = pack8(vals + c);
    }

    // ---- stage V tile (+ dropout noise) transposed: Vt[d][key] ----
    {
      const int row = tid >> 1;
      const int hc  = (tid & 1) * 32;
      const float* sv = V  + (size_t)(j0 + row) * DIM + hc;
      const float* sn = NZ + (size_t)(j0 + row) * DIM + hc;
#pragma unroll
      for (int c = 0; c < 32; c += 4) {
        float4 a = *(const float4*)(sv + c);
        float4 b = *(const float4*)(sn + c);
        Vt[(hc + c + 0) * DIM + row] = (__bf16)(a.x + DROP * b.x);
        Vt[(hc + c + 1) * DIM + row] = (__bf16)(a.y + DROP * b.y);
        Vt[(hc + c + 2) * DIM + row] = (__bf16)(a.z + DROP * b.z);
        Vt[(hc + c + 3) * DIM + row] = (__bf16)(a.w + DROP * b.w);
      }
    }

    // ---- stage RPE window: rel rows [i0-j0+960, +128), bf16 ----
    {
      const int rb = i0 - j0 + 960;     // >= 960 since j0 <= i0
      int gr = rb + tid;
      if (gr > 2 * NSEQ - 2) gr = 2 * NSEQ - 2;   // pad row (never gathered)
      const float* sr = grpe + (size_t)gr * DIM;
#pragma unroll
      for (int c = 0; c < 64; c += 8) {
        float t[8];
        float4 a = *(const float4*)(sr + c);
        float4 b = *(const float4*)(sr + c + 4);
        t[0]=a.x; t[1]=a.y; t[2]=a.z; t[3]=a.w;
        t[4]=b.x; t[5]=b.y; t[6]=b.z; t[7]=b.w;
        *(v8bf*)(Rlds + tid * DIM + c) = pack8(t);
      }
    }

    // prefetch next key tile while this one is consumed
    if (jt + 1 < ntile && tid < 64) {
      const size_t nr = (size_t)(j0 + 64 + tid) * DIM;
      __builtin_prefetch(K  + nr, 0, 1);
      __builtin_prefetch(V  + nr, 0, 1);
      __builtin_prefetch(NZ + nr, 0, 1);
    }
    __syncthreads();

    // ---- QK^T: 16x64 score stripe per wave (f32 accum) ----
    v8f sc[4];
#pragma unroll
    for (int nt = 0; nt < 4; ++nt) {
      const __bf16* kb = Klds + (nt * 16 + lc) * DIM + 16 * hf;
      v16bf b0 = cat16(*(const v8bf*)(kb),      *(const v8bf*)(kb + 8));
      v16bf b1 = cat16(*(const v8bf*)(kb + 32), *(const v8bf*)(kb + 40));
      v8f a = (v8f)0.0f;
      a = WMMA_BF16(qA0, b0, a);
      a = WMMA_BF16(qA1, b1, a);
      sc[nt] = a;
    }

    // ---- RPE bias GEMM: E = Q_wave(16x64) @ Rsub^T -> 16x80 in LDS ----
    float* Ew = Elds + wv * (16 * 81);
#pragma unroll
    for (int et = 0; et < 5; ++et) {
      const __bf16* rbp = Rlds + (wv * 16 + et * 16 + lc) * DIM + 16 * hf;
      v16bf b0 = cat16(*(const v8bf*)(rbp),      *(const v8bf*)(rbp + 8));
      v16bf b1 = cat16(*(const v8bf*)(rbp + 32), *(const v8bf*)(rbp + 40));
      v8f e = (v8f)0.0f;
      e = WMMA_BF16(qA0, b0, e);
      e = WMMA_BF16(qA1, b1, e);
#pragma unroll
      for (int r = 0; r < 8; ++r)
        Ew[(r + 8 * hf) * 81 + et * 16 + lc] = e[r];
    }

    // ---- combine: s = causal * (1 + qk + bias); denom; P as bf16 ----
    __bf16* Pw = Plds + wv * (16 * 64);
    const int iw = i0 + wv * 16;
#pragma unroll
    for (int nt = 0; nt < 4; ++nt) {
      const int jl = nt * 16 + lc;      // column within key tile
      const int jg = j0 + jl;           // global key index
#pragma unroll
      for (int r = 0; r < 8; ++r) {
        const int M = r + 8 * hf;       // row within wave stripe
        float s = sc[nt][r] + Ew[M * 81 + (M - jl + 63)];  // gather band bias
        s = (jg <= iw + M) ? (1.0f + s) : 0.0f;            // causal zero-mask
        dsum[r] += s;
        Pw[M * DIM + jl] = (__bf16)s;
      }
    }

    // ---- P @ V accumulation into output tiles ----
    {
      const __bf16* pb = Pw + lc * DIM + 16 * hf;
      v16bf pA0 = cat16(*(const v8bf*)(pb),      *(const v8bf*)(pb + 8));
      v16bf pA1 = cat16(*(const v8bf*)(pb + 32), *(const v8bf*)(pb + 40));
#pragma unroll
      for (int ct = 0; ct < 4; ++ct) {
        const __bf16* vb = Vt + (ct * 16 + lc) * DIM + 16 * hf;
        v16bf b0 = cat16(*(const v8bf*)(vb),      *(const v8bf*)(vb + 8));
        v16bf b1 = cat16(*(const v8bf*)(vb + 32), *(const v8bf*)(vb + 40));
        accO[ct] = WMMA_BF16(pA0, b0, accO[ct]);
        accO[ct] = WMMA_BF16(pA1, b1, accO[ct]);
      }
    }
    __syncthreads();                    // before restaging K/Vt/Rlds
  }

  // ---- denominator: reduce across each 16-lane half (wave32) ----
#pragma unroll
  for (int m = 1; m < 16; m <<= 1) {
#pragma unroll
    for (int r = 0; r < 8; ++r)
      dsum[r] += __shfl_xor(dsum[r], m);
  }

  // ---- write out = acc / denom ----
#pragma unroll
  for (int ct = 0; ct < 4; ++ct) {
#pragma unroll
    for (int r = 0; r < 8; ++r) {
      const int M = r + 8 * hf;
      O[(size_t)(i0 + wv * 16 + M) * DIM + ct * 16 + lc] = accO[ct][r] / dsum[r];
    }
  }
}

extern "C" void kernel_launch(void* const* d_in, const int* in_sizes, int n_in,
                              void* d_out, int out_size, void* d_ws, size_t ws_size,
                              hipStream_t stream) {
  (void)in_sizes; (void)n_in; (void)out_size; (void)d_ws; (void)ws_size;
  const float* q   = (const float*)d_in[0];
  const float* k   = (const float*)d_in[1];
  const float* v   = (const float*)d_in[2];
  const float* nz  = (const float*)d_in[3];
  const float* rpe = (const float*)d_in[4];
  float* out = (float*)d_out;

  dim3 grid(16, 64);   // 16 row tiles x (b*h)=64 heads
  dim3 block(128);     // 4 waves of 32
  hipLaunchKernelGGL(fastmax_fwd_kernel, grid, block, 0, stream,
                     q, k, v, nz, rpe, out);
}
